// LearnedPrior_74680891343221
// MI455X (gfx1250) — matrix operand order repaired and checked by reading, not compiled, not run on hardware
//
#include <hip/hip_runtime.h>
#include <hip/hip_bf16.h>

typedef __bf16 bf16;
typedef __attribute__((ext_vector_type(16))) __bf16 v16bf;
typedef __attribute__((ext_vector_type(8)))  __bf16 v8bf;
typedef __attribute__((ext_vector_type(8)))  float  v8f;

// global (addrspace(1)) flavors so weight loads lower as global_load, not flat_load
typedef __attribute__((address_space(1))) const bf16 gcbf16;
typedef __attribute__((address_space(1))) const v8bf gcv8bf;

#define DIMZ 512
#define NB   256
#define TT   512
#define NTILE 16          // batch rows per workgroup
#define ROWP 520          // padded LDS row stride (bf16): 1040B = 65*16B -> conflict-free b128
#define THREADS 512       // 16 waves
#define WMAT_ELEMS ((size_t)3 * DIMZ * DIMZ)        // 786432 bf16 per matrix
#define GSTRIDE    ((size_t)32 * 16 * 512)          // packed elems per gate block (32 dtiles)
#define ROWSTRIDE  ((size_t)TT * DIMZ)              // out elems per batch row (1MB bytes)

union Frag { v16bf v; v8bf h[2]; };

__device__ __forceinline__ void ld_a(Frag& f, const bf16* ap) {
    f.h[0] = *(const v8bf*)(ap);        // K = koff..koff+7   -> VGPR 0..3
    f.h[1] = *(const v8bf*)(ap + 16);   // K = 16+koff..+7    -> VGPR 4..7
}
__device__ __forceinline__ void ld_b(Frag& f, gcbf16* bp) {
    f.h[0] = *(const gcv8bf*)(bp);
    f.h[1] = *(const gcv8bf*)(bp + 8);
}
__device__ __forceinline__ v8f wmma(const Frag& a, const Frag& b, v8f c) {
    return __builtin_amdgcn_wmma_f32_16x16x32_bf16(false, a.v, false, b.v,
                                                   (short)0, c, false, false);
}

// Pack W ([3D, D] fp32 row-major) into per-lane B-fragment layout, bf16.
// B tile (K=32 x N=16): lane L -> col L%16, koff = (L>=16)*8; vgpr v -> k = koff+(v&3)*2+(v>>2)*16, +1.
__global__ __launch_bounds__(256) void pack_weights(const float* __restrict__ Wih,
                                                    const float* __restrict__ Whh,
                                                    bf16* __restrict__ pIH,
                                                    bf16* __restrict__ pHH)
{
    const int WORDS = 96 * 16 * 256;          // 32-bit words per matrix
    int idx = blockIdx.x * 256 + threadIdx.x;
    if (idx >= 2 * WORDS) return;
    const float* src = (idx < WORDS) ? Wih : Whh;
    bf16*        dst = (idx < WORDS) ? pIH : pHH;
    int w    = (idx < WORDS) ? idx : idx - WORDS;
    int v    = w & 7;
    int lane = (w >> 3) & 31;
    int kt   = (w >> 8) & 15;
    int jt   = w >> 12;                        // 0..95 gate-column tile
    int row  = jt * 16 + (lane & 15);          // gate-output index (0..1535)
    int kcol = kt * 32 + (lane >> 4) * 8 + (v & 3) * 2 + (v >> 2) * 16;
    dst[(size_t)w * 2]     = (bf16)src[(size_t)row * DIMZ + kcol];
    dst[(size_t)w * 2 + 1] = (bf16)src[(size_t)row * DIMZ + kcol + 1];
}

__global__ __launch_bounds__(THREADS) void gru_recurrent(
    const float* __restrict__ noise, const float* __restrict__ h0,
    const bf16* __restrict__ pIH, const bf16* __restrict__ pHH,
    const float* __restrict__ b_ih, const float* __restrict__ b_hh,
    float* __restrict__ out)
{
    __shared__ bf16 xs[NTILE * ROWP];          // x_t slice, bf16
    __shared__ bf16 hsbuf[2][NTILE * ROWP];    // ping-pong h slice, bf16

    const int tid    = threadIdx.x;
    const int wave   = tid >> 5;
    const int lane   = tid & 31;
    const int lmod   = lane & 15;
    const int lhi    = lane >> 4;              // 0 | 1
    const int wgbase = blockIdx.x * NTILE;     // batch row base

    // A-fragment per-lane addressing (16-bit A 16x32 layout)
    const int a_row_off = lmod * ROWP;
    const int a_k0      = lhi * 8;

    const size_t copystride = (size_t)NB * ROWSTRIDE;

    // persistent fp32 h fragments (C/D layout), 2 d-tiles per wave
    v8f    hstate[2];
    float  brz[2][2], bin[2], bhn[2];
    size_t obase[2];      // out element offset for (v=0) row of this lane's tile
    int    ldsbase[2];    // LDS element offset for (v=0) row of this lane's tile

#pragma unroll
    for (int jt = 0; jt < 2; ++jt) {
        const int dtile = wave * 2 + jt;       // 0..31
        const int col   = dtile * 16 + lmod;
#pragma unroll
        for (int v = 0; v < 8; ++v) {
            const int m = v + lhi * 8;
            float h = h0[(size_t)(wgbase + m) * DIMZ + col];
            hstate[jt][v] = h;
            hsbuf[0][m * ROWP + col] = (bf16)h;
        }
        brz[jt][0] = b_ih[col]            + b_hh[col];
        brz[jt][1] = b_ih[DIMZ + col]     + b_hh[DIMZ + col];
        bin[jt]    = b_ih[2 * DIMZ + col];
        bhn[jt]    = b_hh[2 * DIMZ + col];
        obase[jt]   = (size_t)(wgbase + lhi * 8) * ROWSTRIDE + col;
        ldsbase[jt] = (lhi * 8) * ROWP + col;
    }
    __syncthreads();

    gcbf16* gIH = (gcbf16*)pIH;
    gcbf16* gHH = (gcbf16*)pHH;

    int cur = 0;
    for (int t = 0; t < TT; ++t) {
        // Compiler barrier: pretend the weight bases change each step so the
        // (t-invariant) B-fragment loads are NOT hoisted + spilled to scratch.
        asm volatile("" : "+s"(gIH), "+s"(gHH));

        // ---- stage x_t (fp32 -> bf16) into LDS ----
        const float* xsrc = noise + ((size_t)t * NB + wgbase) * DIMZ;
#pragma unroll
        for (int it = 0; it < (NTILE * DIMZ / 4) / THREADS; ++it) {
            int i   = tid + it * THREADS;      // float4 group index
            int row = i >> 7;                  // 128 groups per row
            int c4  = (i & 127) * 4;
            const float4 f = *(const float4*)(xsrc + (size_t)row * DIMZ + c4);
            bf16* d = &xs[row * ROWP + c4];
            d[0] = (bf16)f.x; d[1] = (bf16)f.y; d[2] = (bf16)f.z; d[3] = (bf16)f.w;
        }
        __syncthreads();

        const bf16* hcur = hsbuf[cur];
        bf16*       hnxt = hsbuf[cur ^ 1];

#pragma unroll
        for (int jt = 0; jt < 2; ++jt) {
            const int dtile = wave * 2 + jt;

            // per-lane packed-weight bases for this dtile (gate g at +g*GSTRIDE)
            gcbf16* whh0 = gHH + (size_t)dtile * (16 * 512) + lane * 16;
            gcbf16* wih0 = gIH + (size_t)dtile * (16 * 512) + lane * 16;

            v8f accr, accz, accin, acchn;
#pragma unroll
            for (int v = 0; v < 8; ++v) {
                accr[v] = brz[jt][0]; accz[v] = brz[jt][1];
                accin[v] = bin[jt];   acchn[v] = bhn[jt];
            }

            // fused K loop: load A_h/A_x once per chunk, 6 B fragments, 6 WMMAs
#pragma unroll
            for (int kt = 0; kt < 16; ++kt) {
                const int aoff = a_row_off + kt * 32 + a_k0;
                Frag ah, ax, b0, b1, b2, b3, b4, b5;
                ld_a(ah, hcur + aoff);
                ld_a(ax, xs   + aoff);
                const size_t ko = (size_t)kt * 512;
                ld_b(b0, whh0 + ko);                 // W_hh, r
                ld_b(b1, whh0 + GSTRIDE + ko);       // W_hh, z
                ld_b(b2, whh0 + 2 * GSTRIDE + ko);   // W_hh, n
                ld_b(b3, wih0 + ko);                 // W_ih, r
                ld_b(b4, wih0 + GSTRIDE + ko);       // W_ih, z
                ld_b(b5, wih0 + 2 * GSTRIDE + ko);   // W_ih, n
                accr  = wmma(ah, b0, accr);
                accz  = wmma(ah, b1, accz);
                acchn = wmma(ah, b2, acchn);
                accr  = wmma(ax, b3, accr);
                accz  = wmma(ax, b4, accz);
                accin = wmma(ax, b5, accin);
            }

            // epilogue: base pointers + immediate offsets (v*ROWSTRIDE fits imm24)
            float* o0 = out + obase[jt] + (size_t)t * DIMZ;
            float* o1 = o0 + copystride;
            float* o2 = o1 + copystride;
            bf16*  hb = hnxt + ldsbase[jt];
#pragma unroll
            for (int v = 0; v < 8; ++v) {
                float r = 1.0f / (1.0f + __expf(-accr[v]));
                float z = 1.0f / (1.0f + __expf(-accz[v]));
                float n = tanhf(accin[v] + r * acchn[v]);
                float hnew = n + z * (hstate[jt][v] - n);
                hstate[jt][v] = hnew;
                hb[v * ROWP]           = (bf16)hnew;
                o0[v * ROWSTRIDE]      = hnew;
                o1[v * ROWSTRIDE]      = hnew;
                o2[v * ROWSTRIDE]      = hnew;
            }
        }
        __syncthreads();   // all reads of hcur/xs done; hnxt fully written
        cur ^= 1;
    }
}

extern "C" void kernel_launch(void* const* d_in, const int* in_sizes, int n_in,
                              void* d_out, int out_size, void* d_ws, size_t ws_size,
                              hipStream_t stream) {
    const float* noise = (const float*)d_in[0];
    const float* h0    = (const float*)d_in[1];
    const float* Wih   = (const float*)d_in[2];
    const float* Whh   = (const float*)d_in[3];
    const float* bih   = (const float*)d_in[4];
    const float* bhh   = (const float*)d_in[5];
    float* out = (float*)d_out;

    bf16* pIH = (bf16*)d_ws;
    bf16* pHH = pIH + WMAT_ELEMS;   // 2 * 1.5MB bf16 weight staging in workspace

    const int total_words = 2 * 96 * 16 * 256;
    pack_weights<<<(total_words + 255) / 256, 256, 0, stream>>>(Wih, Whh, pIH, pHH);
    gru_recurrent<<<NB / NTILE, THREADS, 0, stream>>>(noise, h0, pIH, pHH, bih, bhh, out);
}
